// SHSemanticBiasAttnLayer_91018946937212
// MI455X (gfx1250) — matrix operand order repaired and checked by reading, not compiled, not run on hardware
//
#include <hip/hip_runtime.h>
#include <hip/hip_bf16.h>
#include <math.h>

// ---------------------------------------------------------------- types
typedef __bf16 bf16;
typedef __attribute__((ext_vector_type(16))) __bf16 v16bf;
typedef __attribute__((ext_vector_type(8)))  float  v8f;
typedef __attribute__((ext_vector_type(4)))  unsigned int v4u;

union Frag16 { v16bf v; v4u u[2]; };

// Load a 16-element bf16 WMMA fragment from two 16-byte-aligned halves.
__device__ __forceinline__ v16bf load_frag(const bf16* p0, const bf16* p1) {
    Frag16 f;
    f.u[0] = *(const v4u*)p0;
    f.u[1] = *(const v4u*)p1;
    return f.v;
}

__device__ __forceinline__ v8f wmma_bf16(v16bf a, v16bf b, v8f c) {
    return __builtin_amdgcn_wmma_f32_16x16x32_bf16(
        /*neg_a=*/false, a, /*neg_b=*/false, b,
        /*c_mod=*/(short)0, c, /*reuse_a=*/false, /*reuse_b=*/false);
}

__device__ __forceinline__ float wave_sum16(float s) {
    // sum across the 16-lane half (C/D rows live in 16-lane groups)
    #pragma unroll
    for (int mk = 1; mk < 16; mk <<= 1) s += __shfl_xor(s, mk, 32);
    return s;
}
__device__ __forceinline__ float wave_max16(float s) {
    #pragma unroll
    for (int mk = 1; mk < 16; mk <<= 1) s = fmaxf(s, __shfl_xor(s, mk, 32));
    return s;
}
__device__ __forceinline__ float wave_sum32(float s) {
    #pragma unroll
    for (int mk = 1; mk < 32; mk <<= 1) s += __shfl_xor(s, mk, 32);
    return s;
}

// ---------------------------------------------------------------- problem dims
#define DIM_B   2
#define DIM_C   256
#define DIM_HW  16384          // 128*128
#define DIM_NQ  (DIM_B * DIM_HW)   // 32768 total query rows
#define DIM_NK  4224
#define DIM_NKR (DIM_B * DIM_NK)   // 8448 total key rows
#define DIM_F   1024

// ---------------------------------------------------------------- transpose [b][R][N] -> [b][N][R]
__global__ __launch_bounds__(256) void transpose_kernel(
    const float* __restrict__ in, float* __restrict__ out, int R, int N) {
    __shared__ __align__(16) float tile[32][33];
    const int b = blockIdx.z;
    const float* inb = in + (size_t)b * R * N;
    float* outb = out + (size_t)b * R * N;
    const int x = blockIdx.x * 32 + threadIdx.x;
    #pragma unroll
    for (int j = 0; j < 4; ++j) {
        int y = blockIdx.y * 32 + threadIdx.y + j * 8;
        tile[threadIdx.y + j * 8][threadIdx.x] = inb[(size_t)y * N + x];
    }
    __syncthreads();
    const int x2 = blockIdx.y * 32 + threadIdx.x;
    #pragma unroll
    for (int j = 0; j < 4; ++j) {
        int y2 = blockIdx.x * 32 + threadIdx.y + j * 8;
        outb[(size_t)y2 * R + x2] = tile[threadIdx.x][threadIdx.y + j * 8];
    }
}

// ---------------------------------------------------------------- f32 -> bf16 convert
__global__ __launch_bounds__(256) void cvt_bf16_kernel(
    const float* __restrict__ in, bf16* __restrict__ out, int n4) {
    int i = blockIdx.x * 256 + threadIdx.x;
    int stride = gridDim.x * 256;
    for (; i < n4; i += stride) {
        float4 f = ((const float4*)in)[i];
        out[4 * i + 0] = (bf16)f.x;
        out[4 * i + 1] = (bf16)f.y;
        out[4 * i + 2] = (bf16)f.z;
        out[4 * i + 3] = (bf16)f.w;
    }
}

// ---------------------------------------------------------------- LayerNorm (wave per row, C=256) -> bf16
__global__ __launch_bounds__(256) void ln_kernel(
    const float* __restrict__ x, const float* __restrict__ g,
    const float* __restrict__ b, bf16* __restrict__ out, int rows) {
    const int row = blockIdx.x * 8 + (threadIdx.x >> 5);
    const int lane = threadIdx.x & 31;
    if (row >= rows) return;
    const float* xr = x + (size_t)row * DIM_C;
    float v[8];
    float s = 0.f;
    #pragma unroll
    for (int i = 0; i < 8; ++i) { v[i] = xr[lane + 32 * i]; s += v[i]; }
    const float mu = wave_sum32(s) * (1.f / DIM_C);
    float q = 0.f;
    #pragma unroll
    for (int i = 0; i < 8; ++i) { float d = v[i] - mu; q += d * d; }
    const float rstd = rsqrtf(wave_sum32(q) * (1.f / DIM_C) + 1e-5f);
    bf16* outr = out + (size_t)row * DIM_C;
    #pragma unroll
    for (int i = 0; i < 8; ++i) {
        int c = lane + 32 * i;
        outr[c] = (bf16)((v[i] - mu) * rstd * g[c] + b[c]);
    }
}

// ---------------------------------------------------------------- sem = value @ Wsem + bsem (wave per row)
__global__ __launch_bounds__(256) void sem_kernel(
    const float* __restrict__ value, const float* __restrict__ Wsem,
    const float* __restrict__ bsem, float* __restrict__ sem, int rows) {
    const int row = blockIdx.x * 8 + (threadIdx.x >> 5);
    const int lane = threadIdx.x & 31;
    if (row >= rows) return;
    const float* vr = value + (size_t)row * DIM_C;
    float s = 0.f;
    #pragma unroll
    for (int i = 0; i < 8; ++i) {
        int c = lane + 32 * i;
        s += vr[c] * Wsem[c];
    }
    s = wave_sum32(s);
    if (lane == 0) sem[row] = s + bsem[0];
}

// ---------------------------------------------------------------- generic bf16 WMMA GEMM
//  out[M,N] = act(A[M,K] @ W[K,N] + bias) + residual
//  block = 256 threads (8 waves), tile 128(M) x 64(N), K step 32
template <bool HAS_BIAS, bool HAS_RES, bool GELU_ACT, bool OUT_BF16>
__global__ __launch_bounds__(256) void gemm_kernel(
    const bf16* __restrict__ A, const bf16* __restrict__ W,
    const float* __restrict__ bias, const float* __restrict__ res,
    void* __restrict__ outp, int M, int N, int K) {
    __shared__ __align__(16) bf16 As[128 * 40];   // [row][k] pad->40
    __shared__ __align__(16) bf16 Bs[64 * 40];    // [n][k]  (W transposed) pad->40
    const int tid = threadIdx.x;
    const int lane = tid & 31, w = tid >> 5;
    const int h = lane >> 4, r = lane & 15;
    const int wm = (w & 3) * 32, wn = (w >> 2) * 32;
    const int m0 = blockIdx.y * 128, n0 = blockIdx.x * 64;

    v8f acc[2][2] = {};
    for (int k0 = 0; k0 < K; k0 += 32) {
        __syncthreads();
        { // stage A 128x32 (uint = 2 bf16)
            const unsigned int* Ag = (const unsigned int*)(A + (size_t)m0 * K + k0);
            unsigned int* Au = (unsigned int*)As;
            const int Kru = K >> 1;
            #pragma unroll
            for (int i = 0; i < 8; ++i) {
                int idx = tid + 256 * i;
                int row = idx >> 4, cu = idx & 15;
                Au[row * 20 + cu] = Ag[(size_t)row * Kru + cu];
            }
        }
        { // stage W transposed: Bs[n][kk]
            #pragma unroll
            for (int i = 0; i < 8; ++i) {
                int idx = tid + 256 * i;
                int n = idx & 63, kk = idx >> 6;
                Bs[n * 40 + kk] = W[(size_t)(k0 + kk) * N + n0 + n];
            }
        }
        __syncthreads();
        v16bf af[2], bfr[2];
        #pragma unroll
        for (int mt = 0; mt < 2; ++mt) {
            int row = wm + mt * 16 + r;
            af[mt] = load_frag(&As[row * 40 + 8 * h], &As[row * 40 + 8 * h + 16]);
        }
        #pragma unroll
        for (int nt = 0; nt < 2; ++nt) {
            int n = wn + nt * 16 + r;
            bfr[nt] = load_frag(&Bs[n * 40 + 8 * h], &Bs[n * 40 + 8 * h + 16]);
        }
        #pragma unroll
        for (int mt = 0; mt < 2; ++mt)
            #pragma unroll
            for (int nt = 0; nt < 2; ++nt)
                acc[mt][nt] = wmma_bf16(af[mt], bfr[nt], acc[mt][nt]);
    }
    // epilogue: element e of a tile -> row (e + 8*h), col (lane&15)
    #pragma unroll
    for (int mt = 0; mt < 2; ++mt) {
        #pragma unroll
        for (int nt = 0; nt < 2; ++nt) {
            const int col = n0 + wn + nt * 16 + r;
            float bv = 0.f;
            if (HAS_BIAS) bv = bias[col];
            #pragma unroll
            for (int e = 0; e < 8; ++e) {
                const int row = m0 + wm + mt * 16 + e + 8 * h;
                float val = acc[mt][nt][e] + bv;
                if (GELU_ACT) val = 0.5f * val * (1.f + erff(val * 0.70710678118654752f));
                if (HAS_RES) val += res[(size_t)row * N + col];
                if (OUT_BF16) ((bf16*)outp)[(size_t)row * N + col] = (bf16)val;
                else          ((float*)outp)[(size_t)row * N + col] = val;
            }
        }
    }
}

// ---------------------------------------------------------------- flash attention
// block = 8 waves; wave owns 16 query rows; iterate Nk in tiles of 32 keys
#define KS_STRIDE 264   // 256 + 8 bf16 pad (528B rows, 16B aligned, conflict-free b128)
#define VT_STRIDE 40    // 32 + 8 bf16 pad
__global__ __launch_bounds__(256) void attn_kernel(
    const bf16* __restrict__ q, const bf16* __restrict__ k,
    const bf16* __restrict__ v, const float* __restrict__ sem,
    bf16* __restrict__ out) {
    __shared__ __align__(16) bf16 Ks[32 * KS_STRIDE];       // [key][ch]
    __shared__ __align__(16) bf16 Vt[DIM_C * VT_STRIDE];    // [ch][key]
    __shared__ __align__(16) bf16 Ps[8 * 16 * VT_STRIDE];   // per-wave P[16][32]

    const int tid = threadIdx.x, lane = tid & 31, w = tid >> 5;
    const int h = lane >> 4, r = lane & 15;
    const int qrow0 = blockIdx.x * 128 + w * 16;              // global row in [0, B*Nq)
    const int bidx = (blockIdx.x * 128) / DIM_HW;             // batch
    const bf16* kb = k + (size_t)bidx * DIM_NK * DIM_C;
    const bf16* vb = v + (size_t)bidx * DIM_NK * DIM_C;
    const float* semb = sem + (size_t)bidx * DIM_NK;

    // Q fragments straight from global (A-layout: lane r = row, elems = K-dim)
    v16bf qf[8];
    {
        const bf16* qrow = q + (size_t)(qrow0 + r) * DIM_C;
        #pragma unroll
        for (int kc = 0; kc < 8; ++kc)
            qf[kc] = load_frag(qrow + kc * 32 + 8 * h, qrow + kc * 32 + 8 * h + 16);
    }

    v8f acc[16] = {};
    float mrow[8], lrow[8];
    #pragma unroll
    for (int e = 0; e < 8; ++e) { mrow[e] = -3.0e38f; lrow[e] = 0.f; }

    bf16* Pw = Ps + w * 16 * VT_STRIDE;
    const float scale = 0.0625f;   // C^-0.5

    for (int kt = 0; kt < DIM_NK / 32; ++kt) {
        __syncthreads();
        const bf16* kt_ptr = kb + (size_t)kt * 32 * DIM_C;
        const bf16* vt_ptr = vb + (size_t)kt * 32 * DIM_C;
        if (kt + 1 < DIM_NK / 32) {
            __builtin_prefetch(kt_ptr + 32 * DIM_C + tid * 32, 0, 0);
            __builtin_prefetch(vt_ptr + 32 * DIM_C + tid * 32, 0, 0);
        }
        { // stage K tile row-major (B operand for S = Q @ K^T)
            const v4u* kg = (const v4u*)kt_ptr;
            #pragma unroll
            for (int i = 0; i < 4; ++i) {
                int idx = tid + 256 * i;
                int key = idx >> 5, cu = idx & 31;
                ((v4u*)(Ks + key * KS_STRIDE))[cu] = kg[key * 32 + cu];
            }
        }
        { // stage V tile transposed (B operand for P @ V)
            const unsigned int* vg = (const unsigned int*)vt_ptr;
            #pragma unroll
            for (int i = 0; i < 16; ++i) {
                int idx = tid + 256 * i;             // 4096 uints = 32 keys x 128
                int cu = idx & 127, key = idx >> 7;
                union { unsigned int u; bf16 x[2]; } p;
                p.u = vg[key * 128 + cu];
                Vt[(2 * cu + 0) * VT_STRIDE + key] = p.x[0];
                Vt[(2 * cu + 1) * VT_STRIDE + key] = p.x[1];
            }
        }
        __syncthreads();

        // S[16 x 32] = Q @ K^T   (two 16x16 tiles, K-dim = channels)
        v8f s0 = {}, s1 = {};
        #pragma unroll
        for (int kc = 0; kc < 8; ++kc) {
            v16bf b0 = load_frag(&Ks[r * KS_STRIDE + kc * 32 + 8 * h],
                                 &Ks[r * KS_STRIDE + kc * 32 + 8 * h + 16]);
            v16bf b1 = load_frag(&Ks[(16 + r) * KS_STRIDE + kc * 32 + 8 * h],
                                 &Ks[(16 + r) * KS_STRIDE + kc * 32 + 8 * h + 16]);
            s0 = wmma_bf16(qf[kc], b0, s0);
            s1 = wmma_bf16(qf[kc], b1, s1);
        }

        // scale + semantic bias + online softmax
        const float sem0 = semb[kt * 32 + r];
        const float sem1 = semb[kt * 32 + 16 + r];
        float p0[8], p1[8], sf[8];
        #pragma unroll
        for (int e = 0; e < 8; ++e) {
            float a0 = s0[e] * scale + sem0;
            float a1 = s1[e] * scale + sem1;
            float mloc = wave_max16(fmaxf(a0, a1));
            float mn = fmaxf(mrow[e], mloc);
            sf[e] = __expf(mrow[e] - mn);
            p0[e] = __expf(a0 - mn);
            p1[e] = __expf(a1 - mn);
            mrow[e] = mn;
            float rs = wave_sum16(p0[e] + p1[e]);
            lrow[e] = lrow[e] * sf[e] + rs;
        }
        #pragma unroll
        for (int j = 0; j < 16; ++j)
            #pragma unroll
            for (int e = 0; e < 8; ++e) acc[j][e] *= sf[e];

        // P -> LDS (C-layout -> A-layout transform)
        #pragma unroll
        for (int e = 0; e < 8; ++e) {
            int prow = e + 8 * h;
            Pw[prow * VT_STRIDE + r]      = (bf16)p0[e];
            Pw[prow * VT_STRIDE + 16 + r] = (bf16)p1[e];
        }
        __syncthreads();
        v16bf pf = load_frag(&Pw[r * VT_STRIDE + 8 * h],
                             &Pw[r * VT_STRIDE + 8 * h + 16]);
        // acc += P @ V  (K-dim = 32 keys, 16 output chan tiles)
        #pragma unroll
        for (int j = 0; j < 16; ++j) {
            v16bf bv_ = load_frag(&Vt[(j * 16 + r) * VT_STRIDE + 8 * h],
                                  &Vt[(j * 16 + r) * VT_STRIDE + 8 * h + 16]);
            acc[j] = wmma_bf16(pf, bv_, acc[j]);
        }
    }

    // normalize + store bf16
    #pragma unroll
    for (int e = 0; e < 8; ++e) {
        const float inv = 1.f / lrow[e];
        const int row = qrow0 + e + 8 * h;
        #pragma unroll
        for (int j = 0; j < 16; ++j)
            out[(size_t)row * DIM_C + j * 16 + r] = (bf16)(acc[j][e] * inv);
    }
}

// ---------------------------------------------------------------- host
extern "C" void kernel_launch(void* const* d_in, const int* in_sizes, int n_in,
                              void* d_out, int out_size, void* d_ws, size_t ws_size,
                              hipStream_t stream) {
    const float* query  = (const float*)d_in[0];
    const float* key    = (const float*)d_in[1];
    const float* value  = (const float*)d_in[2];
    const float* ln_q_g = (const float*)d_in[3];
    const float* ln_q_b = (const float*)d_in[4];
    const float* Wq     = (const float*)d_in[5];
    const float* bq     = (const float*)d_in[6];
    const float* Wk     = (const float*)d_in[7];
    const float* Wv     = (const float*)d_in[8];
    const float* bv     = (const float*)d_in[9];
    const float* Wsem   = (const float*)d_in[10];
    const float* bsem   = (const float*)d_in[11];
    const float* Wproj  = (const float*)d_in[12];
    const float* bproj  = (const float*)d_in[13];
    const float* ln_f_g = (const float*)d_in[14];
    const float* ln_f_b = (const float*)d_in[15];
    const float* W1     = (const float*)d_in[16];
    const float* b1     = (const float*)d_in[17];
    const float* W2     = (const float*)d_in[18];
    const float* b2     = (const float*)d_in[19];
    float* out = (float*)d_out;

    // ---- workspace carve-up
    uintptr_t cur = (uintptr_t)d_ws;
    auto carve = [&](size_t bytes) -> void* {
        void* p = (void*)cur;
        cur += (bytes + 255) & ~(size_t)255;
        return p;
    };
    float* x_f32      = (float*)carve((size_t)DIM_NQ * DIM_C * 4);   // also reused for y
    float* x2_f32     = (float*)carve((size_t)DIM_NQ * DIM_C * 4);
    bf16*  xln_bf     = (bf16*) carve((size_t)DIM_NQ * DIM_C * 2);   // LN(x) then LN(x2)
    bf16*  q_bf       = (bf16*) carve((size_t)DIM_NQ * DIM_C * 2);
    bf16*  attn_bf    = (bf16*) carve((size_t)DIM_NQ * DIM_C * 2);
    bf16*  h1_bf      = (bf16*) carve((size_t)DIM_NQ * DIM_F * 2);
    bf16*  key_bf     = (bf16*) carve((size_t)DIM_NKR * DIM_C * 2);
    bf16*  val_bf     = (bf16*) carve((size_t)DIM_NKR * DIM_C * 2);
    bf16*  k_bf       = (bf16*) carve((size_t)DIM_NKR * DIM_C * 2);
    bf16*  v_bf       = (bf16*) carve((size_t)DIM_NKR * DIM_C * 2);
    bf16*  Wq_bf      = (bf16*) carve((size_t)DIM_C * DIM_C * 2);
    bf16*  Wk_bf      = (bf16*) carve((size_t)DIM_C * DIM_C * 2);
    bf16*  Wv_bf      = (bf16*) carve((size_t)DIM_C * DIM_C * 2);
    bf16*  Wproj_bf   = (bf16*) carve((size_t)DIM_C * DIM_C * 2);
    bf16*  W1_bf      = (bf16*) carve((size_t)DIM_C * DIM_F * 2);
    bf16*  W2_bf      = (bf16*) carve((size_t)DIM_F * DIM_C * 2);
    float* sem_f32    = (float*)carve((size_t)DIM_NKR * 4);

    const dim3 tb(32, 8, 1);

    // 1) x[b,n,c] = query[b,c,n]
    transpose_kernel<<<dim3(DIM_HW / 32, DIM_C / 32, DIM_B), tb, 0, stream>>>(
        query, x_f32, DIM_C, DIM_HW);

    // 2) converts
    auto cvt = [&](const float* src, bf16* dst, size_t n) {
        int n4 = (int)(n / 4);
        int blocks = (n4 + 255) / 256;
        if (blocks > 4096) blocks = 4096;
        cvt_bf16_kernel<<<blocks, 256, 0, stream>>>(src, dst, n4);
    };
    cvt(key,   key_bf,   (size_t)DIM_NKR * DIM_C);
    cvt(value, val_bf,   (size_t)DIM_NKR * DIM_C);
    cvt(Wq,    Wq_bf,    (size_t)DIM_C * DIM_C);
    cvt(Wk,    Wk_bf,    (size_t)DIM_C * DIM_C);
    cvt(Wv,    Wv_bf,    (size_t)DIM_C * DIM_C);
    cvt(Wproj, Wproj_bf, (size_t)DIM_C * DIM_C);
    cvt(W1,    W1_bf,    (size_t)DIM_C * DIM_F);
    cvt(W2,    W2_bf,    (size_t)DIM_F * DIM_C);

    // 3) pre-LN for q; semantic bias
    ln_kernel<<<DIM_NQ / 8, 256, 0, stream>>>(x_f32, ln_q_g, ln_q_b, xln_bf, DIM_NQ);
    sem_kernel<<<DIM_NKR / 8, 256, 0, stream>>>(value, Wsem, bsem, sem_f32, DIM_NKR);

    // 4) projections (bf16 WMMA GEMMs)
    gemm_kernel<true,  false, false, true><<<dim3(DIM_C / 64, DIM_NQ / 128), 256, 0, stream>>>(
        xln_bf, Wq_bf, bq, nullptr, q_bf, DIM_NQ, DIM_C, DIM_C);
    gemm_kernel<false, false, false, true><<<dim3(DIM_C / 64, DIM_NKR / 128), 256, 0, stream>>>(
        key_bf, Wk_bf, nullptr, nullptr, k_bf, DIM_NKR, DIM_C, DIM_C);
    gemm_kernel<true,  false, false, true><<<dim3(DIM_C / 64, DIM_NKR / 128), 256, 0, stream>>>(
        val_bf, Wv_bf, bv, nullptr, v_bf, DIM_NKR, DIM_C, DIM_C);

    // 5) flash attention
    attn_kernel<<<DIM_NQ / 128, 256, 0, stream>>>(q_bf, k_bf, v_bf, sem_f32, attn_bf);

    // 6) output projection + residual -> x2 (f32)
    gemm_kernel<true, true, false, false><<<dim3(DIM_C / 64, DIM_NQ / 128), 256, 0, stream>>>(
        attn_bf, Wproj_bf, bproj, x_f32, x2_f32, DIM_NQ, DIM_C, DIM_C);

    // 7) FFN: LN -> W1+GELU -> W2 + residual
    ln_kernel<<<DIM_NQ / 8, 256, 0, stream>>>(x2_f32, ln_f_g, ln_f_b, xln_bf, DIM_NQ);
    gemm_kernel<true, false, true, true><<<dim3(DIM_F / 64, DIM_NQ / 128), 256, 0, stream>>>(
        xln_bf, W1_bf, b1, nullptr, h1_bf, DIM_NQ, DIM_F, DIM_C);
    gemm_kernel<true, true, false, false><<<dim3(DIM_C / 64, DIM_NQ / 128), 256, 0, stream>>>(
        h1_bf, W2_bf, b2, x2_f32, x_f32 /* y reuses x */, DIM_NQ, DIM_C, DIM_F);

    // 8) y[b,n,c] -> out[b,c,n]
    transpose_kernel<<<dim3(DIM_C / 32, DIM_HW / 32, DIM_B), tb, 0, stream>>>(
        x_f32, out, DIM_HW, DIM_C);
    (void)in_sizes; (void)n_in; (void)out_size; (void)ws_size;
}
